// TCN_64579128263365
// MI455X (gfx1250) — matrix operand order repaired and checked by reading, not compiled, not run on hardware
//
#include <hip/hip_runtime.h>
#include <hip/hip_bf16.h>

// ---------------------------------------------------------------------------
// TCN forward for MI455X (gfx1250, wave32, WMMA).
//   Stage 1: causal conv1d(K=3) + bias + relu + scalar projection -> x_output
//            as 3 shifted GEMMs using v_wmma_f32_16x16x32_f16.
//            Barrier-free, double-buffered fragments (loads for step n+1
//            issued ahead of the 8 WMMAs of step n), 2 M-tiles per workgroup
//            so each B fragment feeds 2 WMMAs.
//   Stage 2: global min / mean-square + helper vectors (tiny).
//   Stage 3: masked prefix-product "span" attention, write-once 67MB output.
// ---------------------------------------------------------------------------

typedef __attribute__((ext_vector_type(16))) _Float16 v16h;
typedef __attribute__((ext_vector_type(8)))  float    v8f;

#define S_LEN 1024
#define BATCH 16
#define CIN   512
#define COUT  512
#define KTAP  3
#define NSTEP (KTAP * (CIN / 32))  // 48 K-steps of 32
#define NOUT  (1023 * 1024 * 16)   // attention elements

__device__ __forceinline__ float ht_f(float x) {
    float t = x * 10.0f;
    t = fminf(fmaxf(t, -1.0f), 1.0f);
    return 0.5f * (t + 1.0f);
}

// ---------------------------------------------------------------------------
// Kernel 0: repack conv_w (co,ci,k) f32 -> Wh[k][co][ci] f16 so the GEMM
// K-dimension (ci) is contiguous per output column; zero x_output accumulator.
// ---------------------------------------------------------------------------
__global__ __launch_bounds__(256) void prep_kernel(const float* __restrict__ conv_w,
                                                   _Float16* __restrict__ Wh,
                                                   float* __restrict__ xo_partial) {
    int idx = blockIdx.x * 256 + threadIdx.x;
    if (idx < KTAP * COUT * CIN) {
        int k   = idx / (COUT * CIN);
        int rem = idx - k * (COUT * CIN);
        int co  = rem >> 9;
        int ci  = rem & 511;
        Wh[idx] = (_Float16)conv_w[(co * CIN + ci) * KTAP + k];
    }
    if (idx < S_LEN * BATCH) xo_partial[idx] = 0.0f;
}

// ---------------------------------------------------------------------------
// A-fragment loader: 16-bit A-matrix 16x32, contiguous k order (same as B, not the ISA table order):
// lane (row = lane&15) holds K offsets {offA..offA+15},
// offA = (lane>=16)*16. Loads f32 from x, converts to f16 in-register.
// Rows before the sequence start are zero (causal padding).
// ---------------------------------------------------------------------------
__device__ __forceinline__ v16h load_a_frag(const float* __restrict__ x,
                                            int srow, int row, int cb, int offA) {
    v16h a = {};
    if (srow >= 0) {
        const float* p = x + (size_t)srow * (BATCH * CIN) + row * CIN + cb + offA;
        const float4 c0 = *(const float4*)(p);
        const float4 c1 = *(const float4*)(p + 4);
        const float4 c2 = *(const float4*)(p + 8);
        const float4 c3 = *(const float4*)(p + 12);
        a[0]  = (_Float16)c0.x; a[1]  = (_Float16)c0.y;
        a[2]  = (_Float16)c0.z; a[3]  = (_Float16)c0.w;
        a[4]  = (_Float16)c1.x; a[5]  = (_Float16)c1.y;
        a[6]  = (_Float16)c1.z; a[7]  = (_Float16)c1.w;
        a[8]  = (_Float16)c2.x; a[9]  = (_Float16)c2.y;
        a[10] = (_Float16)c2.z; a[11] = (_Float16)c2.w;
        a[12] = (_Float16)c3.x; a[13] = (_Float16)c3.y;
        a[14] = (_Float16)c3.z; a[15] = (_Float16)c3.w;
    }
    return a;
}

// Load the full fragment set for one K-step: 2 A frags + 4 B frags.
__device__ __forceinline__ void load_step(const float* __restrict__ x,
                                          const _Float16* __restrict__ Wh,
                                          int s0, int row, int offA, int offB,
                                          int wcol, int step,
                                          v16h& a0, v16h& a1, v16h bf[4]) {
    const int k  = step >> 4;           // tap
    const int cb = (step & 15) << 5;    // ci base
    const int srow0 = s0 + k - 2;       // causal shift
    a0 = load_a_frag(x, srow0, row, cb, offA);
    a1 = load_a_frag(x, srow0 + 1, row, cb, offA);
    // B fragments (32x16 f16): lane = column n, 16 contiguous K values at
    // K offset {0|16} per half-wave (dense analogue of sparse 64x16 layout).
    const _Float16* wbase =
        Wh + (size_t)k * (COUT * CIN) + (size_t)(wcol + row) * CIN + cb;
#pragma unroll
    for (int nt = 0; nt < 4; ++nt) {
        bf[nt] = *(const v16h*)(wbase + (size_t)(nt * 16) * CIN + offB);
    }
}

// ---------------------------------------------------------------------------
// Kernel 1: WMMA conv + projection. Barrier-free, double-buffered.
// One workgroup per PAIR of sequence positions (2 M-tiles of 16 batch rows).
// 8 waves, each owns 64 output channels (4 N-tiles of 16).
// ---------------------------------------------------------------------------
__global__ __launch_bounds__(256) void conv_proj_kernel(const float* __restrict__ x,
                                                        const _Float16* __restrict__ Wh,
                                                        const float* __restrict__ conv_b,
                                                        const float* __restrict__ lin_w,
                                                        float* __restrict__ xo_partial) {
    const int s0   = blockIdx.x * 2;      // two sequence positions per WG
    const int t    = threadIdx.x;
    const int lane = t & 31;
    const int w    = t >> 5;              // wave id 0..7
    const int row  = lane & 15;           // M row / N column within tile
    const int hi   = lane >> 4;           // half-wave select
    const int offA = hi << 4;             // A frag K offset: 0 or 16
    const int offB = hi << 4;             // B frag K offset: 0 or 16
    const int wcol = w * 64;              // this wave's first output channel

    v8f acc[2][4];
#pragma unroll
    for (int mt = 0; mt < 2; ++mt)
#pragma unroll
        for (int nt = 0; nt < 4; ++nt) acc[mt][nt] = v8f{};

    v16h A0[2], A1[2], BF[2][4];
    load_step(x, Wh, s0, row, offA, offB, wcol, 0, A0[0], A1[0], BF[0]);

    for (int st = 0; st < NSTEP; st += 2) {
        // Prefetch step st+1 (always <= NSTEP-1), then consume step st.
        load_step(x, Wh, s0, row, offA, offB, wcol, st + 1, A0[1], A1[1], BF[1]);
#pragma unroll
        for (int nt = 0; nt < 4; ++nt) {
            acc[0][nt] = __builtin_amdgcn_wmma_f32_16x16x32_f16(
                false, A0[0], false, BF[0][nt], (short)0, acc[0][nt], false, false);
            acc[1][nt] = __builtin_amdgcn_wmma_f32_16x16x32_f16(
                false, A1[0], false, BF[0][nt], (short)0, acc[1][nt], false, false);
        }
        // Prefetch step st+2 (dummy reload of st on the final trip: keeps
        // control flow scalar-uniform), then consume step st+1.
        const int nxt = (st + 2 < NSTEP) ? st + 2 : st;
        load_step(x, Wh, s0, row, offA, offB, wcol, nxt, A0[0], A1[0], BF[0]);
#pragma unroll
        for (int nt = 0; nt < 4; ++nt) {
            acc[0][nt] = __builtin_amdgcn_wmma_f32_16x16x32_f16(
                false, A0[1], false, BF[1][nt], (short)0, acc[0][nt], false, false);
            acc[1][nt] = __builtin_amdgcn_wmma_f32_16x16x32_f16(
                false, A1[1], false, BF[1][nt], (short)0, acc[1][nt], false, false);
        }
    }

    // Epilogue: per-lane partial of sum_co lin_w[co]*relu(y+conv_b[co])
#pragma unroll
    for (int mt = 0; mt < 2; ++mt) {
        float part[8];
#pragma unroll
        for (int r = 0; r < 8; ++r) part[r] = 0.0f;

#pragma unroll
        for (int nt = 0; nt < 4; ++nt) {
            int n    = wcol + nt * 16 + row;
            float cb = conv_b[n];
            float lw = lin_w[n];
#pragma unroll
            for (int r = 0; r < 8; ++r) {
                float v = acc[mt][nt][r] + cb;
                v = v > 0.0f ? v : 0.0f;
                part[r] += v * lw;
            }
        }

        // Reduce over N (16 lanes within each half-wave), then accumulate.
#pragma unroll
        for (int r = 0; r < 8; ++r) {
            float p = part[r];
            p += __shfl_xor(p, 1, 32);
            p += __shfl_xor(p, 2, 32);
            p += __shfl_xor(p, 4, 32);
            p += __shfl_xor(p, 8, 32);
            if (row == 0) {
                int m = r + (hi ? 8 : 0);   // C/D layout: lanes>=16 hold M=r+8
                atomicAdd(&xo_partial[(s0 + mt) * BATCH + m], p);
            }
        }
    }
}

// ---------------------------------------------------------------------------
// Kernel 2: x_output finalize. min, reg_len, helper vectors, scalar outputs.
// ---------------------------------------------------------------------------
__global__ __launch_bounds__(512) void finalize_kernel(const float* __restrict__ xo_partial,
                                                       const float* __restrict__ lin_b,
                                                       float* __restrict__ xof,
                                                       float* __restrict__ xs,
                                                       float* __restrict__ sq2,
                                                       float* __restrict__ xatt,
                                                       float* __restrict__ out) {
    __shared__ float rmin[512];
    __shared__ float rsum[512];
    const int   t  = threadIdx.x;
    const float lb = lin_b[0];

    float lmin = 3.4e38f, lsum = 0.0f;
    for (int e = 0; e < 32; ++e) {
        int idx = t + e * 512;
        float v = xo_partial[idx] + lb;
        lmin = fminf(lmin, v);
        lsum += v * v;
    }
    rmin[t] = lmin;
    rsum[t] = lsum;
    __syncthreads();
    for (int st = 256; st > 0; st >>= 1) {
        if (t < st) {
            rmin[t] = fminf(rmin[t], rmin[t + st]);
            rsum[t] += rsum[t + st];
        }
        __syncthreads();
    }
    const float minv = rmin[0];
    if (t == 0) {
        out[NOUT]     = 1023.0f;                       // seq_len_data
        out[NOUT + 1] = rsum[0] * (1.0f / 16384.0f);   // reg_len
    }

    for (int e = 0; e < 32; ++e) {
        int idx = t + e * 512;
        float v    = xo_partial[idx] + lb;
        float prev = (idx >= BATCH) ? (xo_partial[idx - BATCH] + lb) : 0.0f;
        xof[idx]  = v;
        xs[idx]   = prev;
        sq2[idx]  = ht_f(prev - v);
        xatt[idx] = v - minv + 10.0f;
    }
}

// ---------------------------------------------------------------------------
// Kernel 3: span attention. One block per query row i (0..1022).
// Thread (chunk c, batch b): 32-j chunk; prefix-product scan across 32 chunks
// via LDS; scores kept in registers so the 67MB output is written exactly once.
// ---------------------------------------------------------------------------
__global__ __launch_bounds__(512) void attn_kernel(const float* __restrict__ xof,
                                                   const float* __restrict__ xs,
                                                   const float* __restrict__ sq2,
                                                   const float* __restrict__ xatt,
                                                   float* __restrict__ out) {
    __shared__ float cp_lds[32][16];
    __shared__ float cs_lds[32][16];

    const int i = blockIdx.x;         // query position, 0..1022
    const int t = threadIdx.x;
    const int b = t & 15;             // batch
    const int c = t >> 4;             // chunk 0..31 (32 j's each)

    const float xoi = xof[i * BATCH + b];

    float f[32];
    float prod = 1.0f;
#pragma unroll
    for (int e = 0; e < 32; ++e) {
        int j = c * 32 + e;
        float fv = 1.0f;
        if (j > i + 1) {
            fv = 1.0f - ht_f(xs[j * BATCH + b] - xoi) * sq2[j * BATCH + b];
        }
        f[e] = fv;
        prod *= fv;
    }
    cp_lds[c][b] = prod;
    __syncthreads();

    float pref = 1.0f;
    for (int cc = 0; cc < c; ++cc) pref *= cp_lds[cc][b];

    float run  = pref;
    float ssum = 0.0f;
#pragma unroll
    for (int e = 0; e < 32; ++e) {
        int j = c * 32 + e;
        run *= f[e];
        float sv = (j > i) ? run * xatt[j * BATCH + b] : 0.0f;
        f[e] = sv;
        ssum += sv;
    }
    cs_lds[c][b] = ssum;
    __syncthreads();

    float tot = 0.0f;
    for (int cc = 0; cc < 32; ++cc) tot += cs_lds[cc][b];
    float inv = 1.0f / tot;

#pragma unroll
    for (int e = 0; e < 32; ++e) {
        int j = c * 32 + e;
        out[(i * 1024 + j) * BATCH + b] = f[e] * inv;
    }
}

// ---------------------------------------------------------------------------
extern "C" void kernel_launch(void* const* d_in, const int* in_sizes, int n_in,
                              void* d_out, int out_size, void* d_ws, size_t ws_size,
                              hipStream_t stream) {
    const float* x      = (const float*)d_in[0];
    const float* conv_w = (const float*)d_in[1];
    const float* conv_b = (const float*)d_in[2];
    const float* lin_w  = (const float*)d_in[3];
    const float* lin_b  = (const float*)d_in[4];
    float* out = (float*)d_out;

    // Workspace layout (~1.9 MB total)
    char* ws = (char*)d_ws;
    _Float16* Wh        = (_Float16*)(ws);             // 3*512*512 f16 = 1.5 MB
    float*    xo_part   = (float*)(ws + 1572864);      // 16384 f32
    float*    xof       = (float*)(ws + 1638400);
    float*    xsh       = (float*)(ws + 1703936);
    float*    sq2       = (float*)(ws + 1769472);
    float*    xatt      = (float*)(ws + 1835008);

    prep_kernel<<<3072, 256, 0, stream>>>(conv_w, Wh, xo_part);
    conv_proj_kernel<<<S_LEN / 2, 256, 0, stream>>>(x, Wh, conv_b, lin_w, xo_part);
    finalize_kernel<<<1, 512, 0, stream>>>(xo_part, lin_b, xof, xsh, sq2, xatt, out);
    attn_kernel<<<1023, 512, 0, stream>>>(xof, xsh, sq2, xatt, out);
}
